// AttentionModel_14568529068734
// MI455X (gfx1250) — matrix operand order repaired and checked by reading, not compiled, not run on hardware
//
#include <hip/hip_runtime.h>

// Problem constants (from reference)
#define BATCH 2048
#define NN    100     // nodes == decode steps
#define DIM   128
#define NH    8
#define DK    16      // DIM / NH
#define MPAD  112     // NN padded to 7 WMMA M-tiles
#define PROJ_STRIDE 385   // 385 mod 64 == 1 -> conflict-free column walks
#define EMB_PITCH 136     // bf16 row pitch, 272B -> 4-bank rotation per row
#define WN_PITCH  136
#define WS_PITCH  264     // W_step^T row pitch (256 + 8), 528B, 16B aligned
#define WO_PITCH  136     // W_out^T  row pitch (128 + 8), 272B, 16B aligned

typedef __bf16  v16bf __attribute__((ext_vector_type(16)));
typedef __bf16  v8bf  __attribute__((ext_vector_type(8)));
typedef float   v8f   __attribute__((ext_vector_type(8)));
typedef float   v4f   __attribute__((ext_vector_type(4)));

// ---- LDS layout (bytes) ----
#define OFF_PROJ   0
#define SZ_PROJ    (MPAD * PROJ_STRIDE * 4)        // 172480: gk|gv|logit_k rows (f32)
#define OFF_WBUF   SZ_PROJ                          // union: wnodeT(bf16) then wstepT+woutT(bf16)
#define SZ_WBUF    (384 * WN_PITCH * 2)             // 104448 >= 128*264*2 + 128*136*2 = 102400
#define OFF_EMBR   (OFF_WBUF + SZ_WBUF)             // 276928
#define SZ_EMBR    (MPAD * EMB_PITCH * 2)           // 30464
#define OFF_GEMB   (OFF_EMBR + SZ_EMBR)             // 307392
#define OFF_FIXED  (OFF_GEMB  + 512)
#define OFF_CTX    (OFF_FIXED + 512)                // 256 floats (16B aligned)
#define OFF_Q      (OFF_CTX   + 1024)
#define OFF_HVEC   (OFF_Q     + 512)
#define OFF_GL     (OFF_HVEC  + 512)
#define OFF_ATTN   (OFF_GL    + 512)                // 8 * 104 floats
#define OFF_LOGITS (OFF_ATTN  + 8*104*4)
#define OFF_MASK   (OFF_LOGITS + 104*4)
#define OFF_MISC   (OFF_MASK  + 104*4)              // ints: first,last,sel
#define OFF_LSE    (OFF_MISC  + 16)
#define SMEM_BYTES (OFF_LSE + 16)                   // 315168 < 320 KB/WGP

__global__ __launch_bounds__(256, 1)
void am_decode_kernel(const float* __restrict__ emb,     // [B,N,D]
                      const float* __restrict__ Wnode,   // [D,3D]
                      const float* __restrict__ Wfixed,  // [D,D]
                      const float* __restrict__ Wstep,   // [2D,D]
                      const float* __restrict__ Wout,    // [D,D]
                      const float* __restrict__ Wph,     // [2D]
                      float* __restrict__ out_logp,      // [B,N,N]
                      float* __restrict__ out_pi)        // [B,N]
{
    extern __shared__ char smem[];
    float*  s_proj   = (float*)(smem + OFF_PROJ);           // row n: [0:128)=gk [128:256)=gv [256:384)=logit_k
    __bf16* s_wnodeT = (__bf16*)(smem + OFF_WBUF);          // [384][WN_PITCH] col-major W_node (phase 2)
    __bf16* s_wstepT = (__bf16*)(smem + OFF_WBUF);          // [128][WS_PITCH] W_step^T (decode)
    __bf16* s_woutT  = (__bf16*)(smem + OFF_WBUF + 128*WS_PITCH*2); // [128][WO_PITCH] W_out^T
    __bf16* s_embR   = (__bf16*)(smem + OFF_EMBR);          // [MPAD][EMB_PITCH] row-major, zero-padded
    float*  s_gemb   = (float*)(smem + OFF_GEMB);
    float*  s_fixed  = (float*)(smem + OFF_FIXED);
    float*  s_ctx    = (float*)(smem + OFF_CTX);
    float*  s_q      = (float*)(smem + OFF_Q);
    float*  s_hvec   = (float*)(smem + OFF_HVEC);
    float*  s_gl     = (float*)(smem + OFF_GL);
    float*  s_attn   = (float*)(smem + OFF_ATTN);
    float*  s_logits = (float*)(smem + OFF_LOGITS);
    int*    s_mask   = (int*)(smem + OFF_MASK);
    int*    s_misc   = (int*)(smem + OFF_MISC);             // [0]=first [1]=last [2]=sel
    float*  s_lse    = (float*)(smem + OFF_LSE);

    const int tid  = threadIdx.x;
    const int lane = tid & 31;
    const int wave = tid >> 5;
    const int b    = blockIdx.x;
    const float* embB = emb + (size_t)b * NN * DIM;

    // ---------- Phase 0: stage emb rows (bf16), W_node^T (bf16), graph mean ----------
    for (int i = tid; i < NN * DIM; i += 256) {             // valid rows: uniform trip count
        int n = i >> 7, k = i & 127;
        s_embR[n * EMB_PITCH + k] = (__bf16)embB[i];
    }
    for (int i = NN * DIM + tid; i < MPAD * DIM; i += 256) { // zero-pad rows 100..111
        int n = i >> 7, k = i & 127;
        s_embR[n * EMB_PITCH + k] = (__bf16)0.0f;
    }
    for (int i = tid; i < DIM * 3 * DIM; i += 256) {        // W_node -> bf16, transposed
        int k = i / (3 * DIM);
        int col = i - k * (3 * DIM);
        s_wnodeT[col * WN_PITCH + k] = (__bf16)Wnode[i];
    }
    if (tid < DIM) {
        float s = 0.f;
        for (int n = 0; n < NN; ++n) s += embB[n * DIM + tid];
        s_gemb[tid] = s * (1.0f / NN);
    }
    if (tid < 104) s_mask[tid] = 0;
    if (tid == 0) { s_misc[0] = 0; s_misc[1] = 0; s_misc[2] = 0; }
    __syncthreads();

    // ---------- Phase 1: fixed_context = graph_emb @ W_fixed (waves 0-3 only) ----------
    if (tid < DIM) {
        float a = 0.f;
        for (int i = 0; i < DIM; ++i) a += s_gemb[i] * Wfixed[i * DIM + tid];
        s_fixed[tid] = a;
    }
    // no barrier needed: phase 2 reads only phase-0 data; waves 4-7 start WMMA now

    // ---------- Phase 2: proj = emb @ W_node via v_wmma_f32_16x16x32_bf16 ----------
    {
        const int half16 = (lane & 16) ? 1 : 0;
        const int ml     = lane & 15;
        for (int tn = wave; tn < 24; tn += 8) {             // 3 n-tiles per wave, uniform
            const int bcol = tn * 16 + ml;
            v16bf Bm[4];
#pragma unroll
            for (int kb = 0; kb < 4; ++kb) {                // hoisted B fragments (7x reuse)
                const int bk0 = kb * 32 + (half16 ? 8 : 0);
                const __bf16* bp = s_wnodeT + bcol * WN_PITCH + bk0;
                v8bf blo = *(const v8bf*)bp;                // ds_load_b128
                v8bf bhi = *(const v8bf*)(bp + 16);         // ds_load_b128
                Bm[kb] = __builtin_shufflevector(blo, bhi,
                          0,1,2,3,4,5,6,7,8,9,10,11,12,13,14,15);
            }
            for (int tm = 0; tm < 7; ++tm) {
                const int ar = tm * 16 + ml;                // padded: always valid
                v8f acc = {};
#pragma unroll
                for (int kb = 0; kb < 4; ++kb) {
                    const int akb = kb * 32 + (half16 ? 8 : 0);
                    const __bf16* ap = s_embR + ar * EMB_PITCH + akb;
                    v8bf alo = *(const v8bf*)ap;            // K = akb .. akb+7
                    v8bf ahi = *(const v8bf*)(ap + 16);     // K = akb+16 .. akb+23
                    v16bf A = __builtin_shufflevector(alo, ahi,
                               0,1,2,3,4,5,6,7,8,9,10,11,12,13,14,15);
                    acc = __builtin_amdgcn_wmma_f32_16x16x32_bf16(
                              false, A, false, Bm[kb], (short)0, acc, false, false);
                }
                const int rbase = tm * 16 + half16 * 8;     // padded proj: unconditional store
                float* cp = s_proj + rbase * PROJ_STRIDE + bcol;
#pragma unroll
                for (int v = 0; v < 8; ++v) cp[v * PROJ_STRIDE] = acc[v];
            }
        }
    }
    __syncthreads();

    // ---------- Phase 2.5: overwrite wnodeT region with transposed bf16 W_step / W_out ----------
    for (int i = tid; i < 256 * 128; i += 256) {            // W_step[i][j] -> wstepT[j][i]
        int k = i >> 7, j = i & 127;
        s_wstepT[j * WS_PITCH + k] = (__bf16)Wstep[i];
    }
    for (int i = tid; i < 128 * 128; i += 256) {            // W_out[i][j] -> woutT[j][i]
        int k = i >> 7, j = i & 127;
        s_woutT[j * WO_PITCH + k] = (__bf16)Wout[i];
    }
    __syncthreads();

    // ---------- Phase 3: sequential greedy decode, 100 steps, all from LDS ----------
    float* outP = out_logp + (size_t)b * NN * NN;
    for (int t = 0; t < NN; ++t) {
        // 3a: step context
        if (t == 0) {
            s_ctx[tid] = Wph[tid];
        } else {
            int node  = (tid < DIM) ? s_misc[0] : s_misc[1]; // first | last
            s_ctx[tid] = embB[node * DIM + (tid & (DIM - 1))];
        }
        __syncthreads();

        // 3b: query = fixed_context + ctx @ W_step  (vector row walk, ds_load_b128)
        if (tid < DIM) {
            float a = s_fixed[tid];
            const __bf16* wr = s_wstepT + tid * WS_PITCH;
            for (int i0 = 0; i0 < 2 * DIM; i0 += 8) {
                v8bf w  = *(const v8bf*)(wr + i0);
                v4f  c0 = *(const v4f*)(s_ctx + i0);
                v4f  c1 = *(const v4f*)(s_ctx + i0 + 4);
#pragma unroll
                for (int e = 0; e < 4; ++e) a += c0[e] * (float)w[e];
#pragma unroll
                for (int e = 0; e < 4; ++e) a += c1[e] * (float)w[e + 4];
            }
            s_q[tid] = a;
        }
        __syncthreads();

        // 3c: per-head masked softmax attention (wave w == head w)
        {
            const int h = wave;
            float qk[DK];
#pragma unroll
            for (int k = 0; k < DK; ++k) qk[k] = s_q[h * DK + k];
            float c[4];
            float mx = -3.0e38f;
#pragma unroll
            for (int j = 0; j < 4; ++j) {
                int n = lane + 32 * j;
                float v = -3.0e38f;
                if (n < NN) {
                    const float* row = s_proj + n * PROJ_STRIDE + h * DK;
                    float d = 0.f;
#pragma unroll
                    for (int k = 0; k < DK; ++k) d += qk[k] * row[k];
                    v = d * 0.25f;                     // 1/sqrt(dk)
                    if (s_mask[n]) v = -1e9f;
                }
                c[j] = v;
                mx = fmaxf(mx, v);
            }
#pragma unroll
            for (int off = 16; off > 0; off >>= 1)
                mx = fmaxf(mx, __shfl_xor(mx, off, 32));
            float sum = 0.f;
#pragma unroll
            for (int j = 0; j < 4; ++j) {
                int n = lane + 32 * j;
                if (n < NN) {
                    float p = __expf(c[j] - mx);
                    s_attn[h * 104 + n] = p;
                    sum += p;
                }
            }
#pragma unroll
            for (int off = 16; off > 0; off >>= 1)
                sum += __shfl_xor(sum, off, 32);
            // weighted sum: lanes [0,16) -> n 0..49, lanes [16,32) -> n 50..99, split by k
            float hv = 0.f;
            int k  = lane & 15;
            int n0 = (lane >> 4) * 50;
            const float* gvcol = s_proj + DIM + h * DK + k;
            const float* aw    = s_attn + h * 104;
            for (int n = n0; n < n0 + 50; ++n)
                hv += aw[n] * gvcol[n * PROJ_STRIDE];
            hv += __shfl_xor(hv, 16, 32);
            if (lane < 16) s_hvec[h * DK + k] = hv / sum;
        }
        __syncthreads();

        // 3d: glimpse = heads @ W_out  (vector row walk, ds_load_b128)
        if (tid < DIM) {
            float a = 0.f;
            const __bf16* wr = s_woutT + tid * WO_PITCH;
            for (int i0 = 0; i0 < DIM; i0 += 8) {
                v8bf w  = *(const v8bf*)(wr + i0);
                v4f  h0 = *(const v4f*)(s_hvec + i0);
                v4f  h1 = *(const v4f*)(s_hvec + i0 + 4);
#pragma unroll
                for (int e = 0; e < 4; ++e) a += h0[e] * (float)w[e];
#pragma unroll
                for (int e = 0; e < 4; ++e) a += h1[e] * (float)w[e + 4];
            }
            s_gl[tid] = a;
        }
        __syncthreads();

        // 3e: tanh-clipped masked logits vs logit_k
        if (tid < NN) {
            const float* lk = s_proj + tid * PROJ_STRIDE + 2 * DIM;
            float a = 0.f;
            for (int i = 0; i < DIM; ++i) a += s_gl[i] * lk[i];
            a = 10.f * tanhf(a * 0.08838834764831845f);   // 1/sqrt(128)
            if (s_mask[tid]) a = -1e9f;
            s_logits[tid] = a;
        }
        __syncthreads();

        // 3f: argmax + log-sum-exp (wave 0)
        if (wave == 0) {
            float m = -3.0e38f; int am = 0x7fffffff;
#pragma unroll
            for (int j = 0; j < 4; ++j) {
                int n = lane + 32 * j;
                if (n < NN) { float v = s_logits[n]; if (v > m) { m = v; am = n; } }
            }
#pragma unroll
            for (int off = 16; off > 0; off >>= 1) {
                float om = __shfl_xor(m,  off, 32);
                int   oa = __shfl_xor(am, off, 32);
                if (om > m || (om == m && oa < am)) { m = om; am = oa; }
            }
            float se = 0.f;
#pragma unroll
            for (int j = 0; j < 4; ++j) {
                int n = lane + 32 * j;
                if (n < NN) se += __expf(s_logits[n] - m);
            }
#pragma unroll
            for (int off = 16; off > 0; off >>= 1) se += __shfl_xor(se, off, 32);
            if (lane == 0) { s_lse[0] = m + __logf(se); s_misc[2] = am; }
        }
        __syncthreads();

        // 3g: emit outputs, update decode state
        float lse = s_lse[0];
        int   sel = s_misc[2];
        if (tid < NN) outP[t * NN + tid] = s_logits[tid] - lse;
        if (tid == 0) {
            s_mask[sel] = 1;
            s_misc[1]   = sel;                 // last
            if (t == 0) s_misc[0] = sel;       // first
            out_pi[(size_t)b * NN + t] = (float)sel;
        }
        __syncthreads();
    }
}

extern "C" void kernel_launch(void* const* d_in, const int* in_sizes, int n_in,
                              void* d_out, int out_size, void* d_ws, size_t ws_size,
                              hipStream_t stream) {
    (void)in_sizes; (void)n_in; (void)d_ws; (void)ws_size; (void)out_size;
    const float* emb    = (const float*)d_in[0];
    const float* Wnode  = (const float*)d_in[1];
    const float* Wfixed = (const float*)d_in[2];
    const float* Wstep  = (const float*)d_in[3];
    const float* Wout   = (const float*)d_in[4];
    const float* Wph    = (const float*)d_in[5];
    float* out_logp = (float*)d_out;                         // [B,N,N]
    float* out_pi   = out_logp + (size_t)BATCH * NN * NN;    // [B,N] (indices as floats)

    (void)hipFuncSetAttribute((const void*)am_decode_kernel,
                              hipFuncAttributeMaxDynamicSharedMemorySize, SMEM_BYTES);
    am_decode_kernel<<<BATCH, 256, SMEM_BYTES, stream>>>(
        emb, Wnode, Wfixed, Wstep, Wout, Wph, out_logp, out_pi);
}
